// Attention_26018911879178
// MI455X (gfx1250) — matrix operand order repaired and checked by reading, not compiled, not run on hardware
//
#include <hip/hip_runtime.h>
#include <math.h>

// ---------------------------------------------------------------------------
// MHA forward for B=2, T=2048, C=1024, H=16, D=64 on gfx1250 (MI455X).
//   1) cvt fp32->bf16: x -> Xb, Wq/Wk/Wv/Wo -> *_b
//   2) gemm_xwt (LDS-staged, async double-buffered): Q/K/V projections
//      (V stored transposed [B,H,D,T] so PV B-fragments are contiguous)
//   3) flash_attn: online-softmax attention, O (bf16) overwrites Xb
//   4) gemm_xwt: d_out = O @ Wo^T (fp32)
// Matrix ops: v_wmma_f32_16x16x32_bf16. B tiles staged with
// global_load_async_to_lds_b128 + s_wait_asynccnt (ASYNCcnt path).
// ---------------------------------------------------------------------------

typedef __bf16 bf16_t;
typedef bf16_t v16bf __attribute__((ext_vector_type(16)));
typedef bf16_t v8bf  __attribute__((ext_vector_type(8)));
typedef bf16_t v4bf  __attribute__((ext_vector_type(4)));
typedef float  v8f   __attribute__((ext_vector_type(8)));
typedef float  v4f   __attribute__((ext_vector_type(4)));

#define DEVI static __device__ __forceinline__

constexpr int Bc = 2;
constexpr int Tc = 2048;
constexpr int Cc = 1024;
constexpr int Hc = 16;
constexpr int Dc = 64;
constexpr int BT = Bc * Tc;       // 4096 rows

DEVI v8bf ld8(const bf16_t* p) { return *(const v8bf*)p; }

DEVI v16bf combine(v8bf lo, v8bf hi) {
  return __builtin_shufflevector(lo, hi, 0, 1, 2, 3, 4, 5, 6, 7,
                                         8, 9, 10, 11, 12, 13, 14, 15);
}

DEVI v8f wmma_bf16(v16bf a, v16bf b, v8f c) {
  // (neg_a, A, neg_b, B, c_mod, C, reuse_a, reuse_b)
  return __builtin_amdgcn_wmma_f32_16x16x32_bf16(false, a, false, b,
                                                 (short)0, c, false, false);
}

// Async 16B/lane copy global -> LDS (ASYNCcnt-tracked).
// lds_off = byte offset within the wave's LDS allocation (low 32 bits of the
// generic shared-memory address per the flat-aperture rules).
DEVI void async_ld16(unsigned int lds_off, const void* gptr) {
  asm volatile("global_load_async_to_lds_b128 %0, %1, off"
               :: "v"(lds_off), "v"((unsigned long long)(uintptr_t)gptr)
               : "memory");
}
DEVI void wait_async0() { asm volatile("s_wait_asynccnt 0x0" ::: "memory"); }

// ---------------------------------------------------------------------------
// fp32 -> bf16 conversion (vectorized x4), n4 = n/4 threads
// ---------------------------------------------------------------------------
__global__ void cvt_f32_bf16(const float* __restrict__ in,
                             bf16_t* __restrict__ out, int n4) {
  int i = blockIdx.x * blockDim.x + threadIdx.x;
  if (i >= n4) return;
  v4f x = *(const v4f*)(in + 4 * (size_t)i);
  v4bf y;
  y[0] = (bf16_t)x[0];
  y[1] = (bf16_t)x[1];
  y[2] = (bf16_t)x[2];
  y[3] = (bf16_t)x[3];
  *(v4bf*)(out + 4 * (size_t)i) = y;
}

// ---------------------------------------------------------------------------
// Y = alpha * (A @ W^T), A:[M,Cc] bf16 row-major, W:[Cc,Cc] bf16 row-major.
// Block tile: 128 rows x 64 cols (8 waves x 16 rows, shared column group).
// The 64x32 B tile per K-step is staged cooperatively into LDS with async
// loads (double-buffered); A fragments are double-buffered in registers.
// mode 0: store bf16 [M,Cc]
// mode 1: store bf16 transposed head-wise to [B,H,D,T]  (for V)
// mode 2: store fp32 [M,Cc]                             (final output)
// ---------------------------------------------------------------------------
__global__ __launch_bounds__(256) void gemm_xwt(const bf16_t* __restrict__ A,
                                                const bf16_t* __restrict__ W,
                                                void* __restrict__ outp,
                                                int M, float alpha, int mode) {
  __shared__ alignas(16) bf16_t Bs[2][64][32];   // 2 x 4KB double buffer
  const int lane = threadIdx.x & 31;
  const int wIn  = threadIdx.x >> 5;
  const int ng = blockIdx.x & (Cc / 64 - 1);     // 16 column groups
  const int bm = blockIdx.x >> 4;
  const int m0 = bm * 128 + wIn * 16;
  const int n0 = ng * 64;
  const int half = lane >> 4, l15 = lane & 15;
  if (m0 >= M) return;                            // never taken (exact grid)

  // Cooperative staging map: thread t copies 8 bf16 of W row (n0 + t/4),
  // k-offset (t%4)*8, one b128 async copy per thread per K-step.
  const int tid = threadIdx.x;
  const int srow = tid >> 2, skoff = (tid & 3) * 8;
  const bf16_t* sgp = W + (size_t)(n0 + srow) * Cc + skoff;
  const unsigned int lds0 = (unsigned int)(uintptr_t)&Bs[0][srow][skoff];
  const unsigned int lds1 = (unsigned int)(uintptr_t)&Bs[1][srow][skoff];

  // A fragment: lane holds row m0+l15; K chunks at +8*half and +16+8*half.
  const bf16_t* arow = A + (size_t)(m0 + l15) * Cc + 8 * half;

  v8f c0 = {}, c1 = {}, c2 = {}, c3 = {};

  async_ld16(lds0, sgp);                          // stage K-block 0 -> buf 0
  v16bf a_cur = combine(ld8(arow), ld8(arow + 16));

  const int NK = Cc / 32;                         // 32 K-steps
  for (int kt = 0; kt < NK; ++kt) {
    const int cur = kt & 1;
    wait_async0();        // our stage for buf `cur` has landed
    __syncthreads();      // all waves' stages landed; prev readers drained
    if (kt + 1 < NK) {    // stage next K-block into the other buffer
      async_ld16(cur ? lds0 : lds1, sgp + (kt + 1) * 32);
    }
    // Issue next A fragment loads before consuming the current one.
    v16bf a_nxt = a_cur;
    if (kt + 1 < NK) {
      const bf16_t* ap = arow + (kt + 1) * 32;
      __builtin_prefetch(ap + 32, 0, 3);
      a_nxt = combine(ld8(ap), ld8(ap + 16));
    }
    // B fragments from LDS: lane = column n, K chunks at +8*half and +16+8*half.
    v16bf b0 = combine(*(const v8bf*)&Bs[cur][ 0 + l15][8 * half],
                       *(const v8bf*)&Bs[cur][ 0 + l15][16 + 8 * half]);
    v16bf b1 = combine(*(const v8bf*)&Bs[cur][16 + l15][8 * half],
                       *(const v8bf*)&Bs[cur][16 + l15][16 + 8 * half]);
    v16bf b2 = combine(*(const v8bf*)&Bs[cur][32 + l15][8 * half],
                       *(const v8bf*)&Bs[cur][32 + l15][16 + 8 * half]);
    v16bf b3 = combine(*(const v8bf*)&Bs[cur][48 + l15][8 * half],
                       *(const v8bf*)&Bs[cur][48 + l15][16 + 8 * half]);
    c0 = wmma_bf16(a_cur, b0, c0);
    c1 = wmma_bf16(a_cur, b1, c1);
    c2 = wmma_bf16(a_cur, b2, c2);
    c3 = wmma_bf16(a_cur, b3, c3);
    a_cur = a_nxt;
  }

#pragma unroll
  for (int nt = 0; nt < 4; ++nt) {
    v8f c = (nt == 0) ? c0 : (nt == 1) ? c1 : (nt == 2) ? c2 : c3;
#pragma unroll
    for (int j = 0; j < 8; ++j) {
      // C/D layout: lanes 0-15 hold rows j, lanes 16-31 rows j+8; N = l15.
      int row = m0 + j + 8 * half;
      int col = n0 + nt * 16 + l15;
      float v = alpha * c[j];
      if (mode == 0) {
        ((bf16_t*)outp)[(size_t)row * Cc + col] = (bf16_t)v;
      } else if (mode == 1) {
        int bb = row >> 11;          // token / T
        int t  = row & (Tc - 1);
        int h  = col >> 6;           // col / D
        int d  = col & (Dc - 1);
        ((bf16_t*)outp)[(((size_t)(bb * Hc + h)) * Dc + d) * Tc + t] = (bf16_t)v;
      } else {
        ((float*)outp)[(size_t)row * Cc + col] = v;
      }
    }
  }
}

// ---------------------------------------------------------------------------
// Flash attention: one wave per 16-query tile per (b,h); 32 keys per step.
// Qb,Kb: [B*T, C] bf16 (head-interleaved), Vt: [B,H,D,T] bf16.
// Scale 1/sqrt(D) was folded into Q. Causal mask computed analytically.
// ---------------------------------------------------------------------------
__global__ __launch_bounds__(256) void flash_attn(const bf16_t* __restrict__ Qb,
                                                  const bf16_t* __restrict__ Kb,
                                                  const bf16_t* __restrict__ Vt,
                                                  bf16_t* __restrict__ Ob) {
  __shared__ float Pbuf[8][16 * 32];     // per-wave 16x32 P tile (fp32)
  const int lane = threadIdx.x & 31;
  const int wIn  = threadIdx.x >> 5;
  const int wave = (blockIdx.x * blockDim.x + threadIdx.x) >> 5;
  const int QT = Tc / 16;                // 128 query tiles per (b,h)
  const int qt = wave % QT;
  const int h  = (wave / QT) % Hc;
  const int b  = wave / (QT * Hc);
  const int q0 = qt * 16;
  const int half = lane >> 4, l15 = lane & 15;
  float* P = Pbuf[wIn];

  // Q A-fragments for d=[0,32) and d=[32,64), loaded once.
  const bf16_t* qrow = Qb + ((size_t)(b * Tc) + q0 + l15) * Cc + h * Dc;
  v16bf aq0 = combine(ld8(qrow + 8 * half),      ld8(qrow + 16 + 8 * half));
  v16bf aq1 = combine(ld8(qrow + 32 + 8 * half), ld8(qrow + 48 + 8 * half));

  const bf16_t* krow = Kb + ((size_t)(b * Tc) + l15) * Cc + h * Dc;
  const bf16_t* vcol = Vt + (((size_t)(b * Hc + h)) * Dc + l15) * Tc + 8 * half;

  v8f o0 = {}, o1 = {}, o2 = {}, o3 = {};
  float m_run[8], l_run[8];
#pragma unroll
  for (int j = 0; j < 8; ++j) { m_run[j] = -3.0e38f; l_run[j] = 0.0f; }

  const int kbEnd = (q0 + 15) / 32;
  for (int kb = 0; kb <= kbEnd; ++kb) {
    const int kt0 = kb * 32, kt1 = kb * 32 + 16;
    // ---- S = Q @ K^T for two 16-key tiles ----
    v8f s0 = {}, s1 = {};
    {
      const bf16_t* kp = krow + (size_t)kt0 * Cc;
      v16bf b0 = combine(ld8(kp + 8 * half),      ld8(kp + 16 + 8 * half));
      v16bf b1 = combine(ld8(kp + 32 + 8 * half), ld8(kp + 48 + 8 * half));
      s0 = wmma_bf16(aq0, b0, s0);
      s0 = wmma_bf16(aq1, b1, s0);
      kp += (size_t)16 * Cc;
      v16bf b2 = combine(ld8(kp + 8 * half),      ld8(kp + 16 + 8 * half));
      v16bf b3 = combine(ld8(kp + 32 + 8 * half), ld8(kp + 48 + 8 * half));
      s1 = wmma_bf16(aq0, b2, s1);
      s1 = wmma_bf16(aq1, b3, s1);
    }
    // ---- issue V fragment loads now: independent of the softmax below ----
    const bf16_t* vp = vcol + kb * 32;
    v16bf bv0 = combine(ld8(vp),           ld8(vp + 16));
    v16bf bv1 = combine(ld8(vp + 16 * Tc), ld8(vp + 16 * Tc + 16));
    v16bf bv2 = combine(ld8(vp + 32 * Tc), ld8(vp + 32 * Tc + 16));
    v16bf bv3 = combine(ld8(vp + 48 * Tc), ld8(vp + 48 * Tc + 16));
    // ---- causal mask ----
#pragma unroll
    for (int j = 0; j < 8; ++j) {
      int rg = q0 + j + 8 * half;
      if (kt0 + l15 > rg) s0[j] = -1.0e30f;
      if (kt1 + l15 > rg) s1[j] = -1.0e30f;
    }
    // ---- online softmax (row = j + 8*half; N spans the 16-lane half) ----
    float scl[8];
#pragma unroll
    for (int j = 0; j < 8; ++j) {
      float v = fmaxf(s0[j], s1[j]);
      v = fmaxf(v, __shfl_xor(v, 1, 32));
      v = fmaxf(v, __shfl_xor(v, 2, 32));
      v = fmaxf(v, __shfl_xor(v, 4, 32));
      v = fmaxf(v, __shfl_xor(v, 8, 32));
      float mNew = fmaxf(m_run[j], v);
      float p0 = __expf(s0[j] - mNew);
      float p1 = __expf(s1[j] - mNew);
      float rs = p0 + p1;
      rs += __shfl_xor(rs, 1, 32);
      rs += __shfl_xor(rs, 2, 32);
      rs += __shfl_xor(rs, 4, 32);
      rs += __shfl_xor(rs, 8, 32);
      scl[j]   = __expf(m_run[j] - mNew);
      l_run[j] = l_run[j] * scl[j] + rs;
      m_run[j] = mNew;
      int r = j + 8 * half;
      P[r * 32 + l15]      = p0;     // C-layout -> row-major LDS
      P[r * 32 + 16 + l15] = p1;
    }
#pragma unroll
    for (int j = 0; j < 8; ++j) {
      o0[j] *= scl[j]; o1[j] *= scl[j]; o2[j] *= scl[j]; o3[j] *= scl[j];
    }

    __builtin_amdgcn_wave_barrier();
    // LDS ops from the same wave are in-order; ensure the P stores landed.
    asm volatile("s_wait_dscnt 0" ::: "memory");

    // ---- re-read P as a 16x32 A-fragment, cast to bf16 ----
    const float* pr = P + l15 * 32;
    v4f pA = *(const v4f*)(pr + 8 * half);
    v4f pB = *(const v4f*)(pr + 8 * half + 4);
    v4f pC = *(const v4f*)(pr + 16 + 8 * half);
    v4f pD = *(const v4f*)(pr + 16 + 8 * half + 4);
    v16bf ap;
#pragma unroll
    for (int i = 0; i < 4; ++i) {
      ap[i]      = (bf16_t)pA[i];
      ap[4 + i]  = (bf16_t)pB[i];
      ap[8 + i]  = (bf16_t)pC[i];
      ap[12 + i] = (bf16_t)pD[i];
    }

    // ---- O += P @ V ----
    o0 = wmma_bf16(ap, bv0, o0);
    o1 = wmma_bf16(ap, bv1, o1);
    o2 = wmma_bf16(ap, bv2, o2);
    o3 = wmma_bf16(ap, bv3, o3);
  }

  // ---- epilogue: normalize and store O (bf16, head-interleaved) ----
#pragma unroll
  for (int j = 0; j < 8; ++j) {
    float inv = 1.0f / l_run[j];
    size_t base = ((size_t)(b * Tc) + q0 + j + 8 * half) * Cc + h * Dc + l15;
    Ob[base]      = (bf16_t)(o0[j] * inv);
    Ob[base + 16] = (bf16_t)(o1[j] * inv);
    Ob[base + 32] = (bf16_t)(o2[j] * inv);
    Ob[base + 48] = (bf16_t)(o3[j] * inv);
  }
}

// ---------------------------------------------------------------------------
// Host launcher
// ---------------------------------------------------------------------------
extern "C" void kernel_launch(void* const* d_in, const int* in_sizes, int n_in,
                              void* d_out, int out_size, void* d_ws, size_t ws_size,
                              hipStream_t stream) {
  const float* x  = (const float*)d_in[0];
  // d_in[1] is the causal mask (unused; mask computed analytically)
  const float* Wq = (const float*)d_in[2];
  const float* Wk = (const float*)d_in[3];
  const float* Wv = (const float*)d_in[4];
  const float* Wo = (const float*)d_in[5];

  char* ws = (char*)d_ws;
  const size_t MB = 1024 * 1024;
  bf16_t* Xb  = (bf16_t*)(ws + 0 * MB);   // 8 MB; reused as O after attention
  bf16_t* Wqb = (bf16_t*)(ws + 8 * MB);   // 2 MB each
  bf16_t* Wkb = (bf16_t*)(ws + 10 * MB);
  bf16_t* Wvb = (bf16_t*)(ws + 12 * MB);
  bf16_t* Wob = (bf16_t*)(ws + 14 * MB);
  bf16_t* Qb  = (bf16_t*)(ws + 16 * MB);  // 8 MB
  bf16_t* Kb  = (bf16_t*)(ws + 24 * MB);  // 8 MB
  bf16_t* Vtb = (bf16_t*)(ws + 32 * MB);  // 8 MB  -> total 40 MB

  // 1) conversions
  {
    int n4 = (BT * Cc) / 4;               // 1,048,576
    cvt_f32_bf16<<<n4 / 256, 256, 0, stream>>>(x, Xb, n4);
    int w4 = (Cc * Cc) / 4;               // 262,144
    cvt_f32_bf16<<<w4 / 256, 256, 0, stream>>>(Wq, Wqb, w4);
    cvt_f32_bf16<<<w4 / 256, 256, 0, stream>>>(Wk, Wkb, w4);
    cvt_f32_bf16<<<w4 / 256, 256, 0, stream>>>(Wv, Wvb, w4);
    cvt_f32_bf16<<<w4 / 256, 256, 0, stream>>>(Wo, Wob, w4);
  }

  // 2) projections: (BT/128) x (Cc/64) = 32 x 16 = 512 blocks x 256 thr
  const int gemmBlocks = (BT / 128) * (Cc / 64);  // 512
  const float scale = 0.125f;  // 1/sqrt(64)
  gemm_xwt<<<gemmBlocks, 256, 0, stream>>>(Xb, Wqb, Qb,  BT, scale, 0);
  gemm_xwt<<<gemmBlocks, 256, 0, stream>>>(Xb, Wkb, Kb,  BT, 1.0f, 0);
  gemm_xwt<<<gemmBlocks, 256, 0, stream>>>(Xb, Wvb, Vtb, BT, 1.0f, 1);

  // 3) attention: waves = B*H*(T/16) = 4096 -> 512 blocks x 256 thr
  flash_attn<<<(Bc * Hc * (Tc / 16)) * 32 / 256, 256, 0, stream>>>(Qb, Kb, Vtb, Xb);

  // 4) output projection -> fp32 d_out
  gemm_xwt<<<gemmBlocks, 256, 0, stream>>>(Xb, Wob, d_out, BT, 1.0f, 2);
}
